// OptimizedAttention_46385646797336
// MI455X (gfx1250) — matrix operand order repaired and checked by reading, not compiled, not run on hardware
//
#include <hip/hip_runtime.h>
#include <hip/hip_bf16.h>

// ---------------------------------------------------------------------------
// Attention block on gfx1250 (MI455X): bf16 WMMA + TDM(LDS) double buffering.
// ---------------------------------------------------------------------------

typedef __bf16 bf16_t;
typedef __attribute__((ext_vector_type(16))) __bf16 v16bf;
typedef __attribute__((ext_vector_type(8)))  float  v8f;
typedef __attribute__((ext_vector_type(4)))  unsigned int v4u;
typedef __attribute__((ext_vector_type(8)))  int v8i;
typedef __attribute__((ext_vector_type(4)))  int v4i;

#define ATT_B 4
#define ATT_T 2048
#define ATT_C 2048
#define ATT_H 16
#define ATT_D 128

#if __has_builtin(__builtin_amdgcn_tensor_load_to_lds)
#define USE_TDM 1
#else
#define USE_TDM 0
#endif

// ---- fragment loaders for CDNA5 WMMA: contiguous k order, same for A and B -------
// A-matrix 16x32 bf16, row-major src, leading dim ld (elements).
// lane L: row = L%16; holds K = c0..c0+15 contiguous, c0 = (L<16)?0:16.
__device__ __forceinline__ v16bf load_a_frag(const bf16_t* src, int ld, int lane) {
    int row = lane & 15;
    int c0  = (lane < 16) ? 0 : 16;
    const bf16_t* p = src + (size_t)row * ld + c0;
    v16bf a;
    ((uint4*)&a)[0] = *((const uint4*)p);        // K = c0..c0+7
    ((uint4*)&a)[1] = *((const uint4*)(p + 8));  // K = c0+8..c0+15
    return a;
}

// B-matrix 32x16 bf16 from [N][K] row-major (contiguous along K).
// lane L: col = L%16; holds K = k0..k0+15 contiguous, k0 = (L<16)?0:16.
__device__ __forceinline__ v16bf load_b_frag(const bf16_t* src, int ld, int lane) {
    int col = lane & 15;
    int k0  = (lane < 16) ? 0 : 16;
    const bf16_t* p = src + (size_t)col * ld + k0;
    v16bf b;
    ((uint4*)&b)[0] = *((const uint4*)p);
    ((uint4*)&b)[1] = *((const uint4*)(p + 8));
    return b;
}

#define WMMA_BF16(a, b, c) \
    __builtin_amdgcn_wmma_f32_16x16x32_bf16(false, (a), false, (b), (short)0, (c), false, false)

// ---- Tensor Data Mover: 2D tile load (64 rows x 64 bf16) into LDS ---------
// D# group0/group1 bit layout per CDNA5 ISA.
// 6-arg builtin form: (v4u g0, v8i g1, v4i g2, v4i g3, v8i extra, i32 cpol).
#if USE_TDM
__device__ __forceinline__ void tdm_load_2d(unsigned lds_off, const bf16_t* gsrc,
                                            unsigned tile_d0, unsigned tile_d1,
                                            unsigned stride_elems) {
    unsigned long long ga = (unsigned long long)gsrc;
    v4u g0;
    g0[0] = 1u;                                   // count=1 (valid), no gather
    g0[1] = lds_off;                              // lds_addr (bytes)
    g0[2] = (unsigned)ga;                         // global_addr[31:0]
    g0[3] = (unsigned)(ga >> 32) | 0x80000000u;   // global_addr[56:32] | type=2
    v8i g1;
    g1[0] = (int)(1u << 16);                      // data_size=1 (2B), mask=0
    g1[1] = (int)((tile_d0 & 0xFFFFu) << 16);     // tensor_dim0 = tile_d0
    g1[2] = (int)((tile_d0 >> 16) | ((tile_d1 & 0xFFFFu) << 16)); // tensor_dim1
    g1[3] = (int)((tile_d1 >> 16) | (tile_d0 << 16));             // tile_dim0
    g1[4] = (int)(tile_d1 & 0xFFFFu);             // tile_dim1 ; tile_dim2=0
    g1[5] = (int)stride_elems;                    // tensor_dim0_stride low
    g1[6] = 0;                                    // stride high / dim1_stride
    g1[7] = 0;
    v4i z4 = {0, 0, 0, 0};
    v8i z8 = {0, 0, 0, 0, 0, 0, 0, 0};
    __builtin_amdgcn_tensor_load_to_lds(g0, g1, z4, z4, z8, 0);
}
__device__ __forceinline__ void tdm_wait() {
#if __has_builtin(__builtin_amdgcn_s_wait_tensorcnt)
    __builtin_amdgcn_s_wait_tensorcnt(0);
#else
    __asm__ volatile("s_wait_tensorcnt 0x0" ::: "memory");
#endif
}
#endif

// ---------------------------------------------------------------------------
// Kernel 1: fp32 -> bf16 conversion (grid-stride)
// ---------------------------------------------------------------------------
__global__ void k_f32_to_bf16(const float* __restrict__ in,
                              bf16_t* __restrict__ out, size_t n) {
    size_t i = (size_t)blockIdx.x * blockDim.x + threadIdx.x;
    size_t stride = (size_t)gridDim.x * blockDim.x;
    for (; i < n; i += stride) out[i] = (bf16_t)in[i];
}

// ---------------------------------------------------------------------------
// Kernel 2/5: WMMA GEMM  out[M][N] = A[M][K] * B[N][K]^T
// Block = 4 waves, 64x64 block tile, K-step 64, TDM-fed LDS double buffer.
// ---------------------------------------------------------------------------
#define GK 64   // K-step
__global__ void __launch_bounds__(128)
k_wmma_gemm(const bf16_t* __restrict__ A, const bf16_t* __restrict__ Bm,
            bf16_t* __restrict__ outb, float* __restrict__ outf,
            int M, int N, int K) {
    __shared__ bf16_t ldsA[2][64 * GK];   // 8 KB per buffer
    __shared__ bf16_t ldsB[2][64 * GK];

    int lane = threadIdx.x & 31;
    int wave = threadIdx.x >> 5;
    int tilesN = N >> 6;
    int bm = blockIdx.x / tilesN;
    int bn = blockIdx.x % tilesN;

    const bf16_t* Ablk = A  + (size_t)bm * 64 * K;
    const bf16_t* Bblk = Bm + (size_t)bn * 64 * K;

#if USE_TDM
    unsigned la0 = (unsigned)(size_t)&ldsA[0][0];
    unsigned lb0 = (unsigned)(size_t)&ldsB[0][0];
    const unsigned bufBytes = 64 * GK * 2;
    // prologue: stage k-step 0
    if (wave == 0) tdm_load_2d(la0, Ablk, GK, 64, (unsigned)K);
    if (wave == 1) tdm_load_2d(lb0, Bblk, GK, 64, (unsigned)K);
    tdm_wait();
    __syncthreads();
#endif

    v8f acc[4] = {v8f{0}, v8f{0}, v8f{0}, v8f{0}};
    int nsteps = K / GK;

    for (int s = 0; s < nsteps; ++s) {
        int buf = s & 1;
#if USE_TDM
        if (s + 1 < nsteps) {   // kick off next panels while computing
            if (wave == 0)
                tdm_load_2d(la0 + (buf ^ 1) * bufBytes, Ablk + (s + 1) * GK,
                            GK, 64, (unsigned)K);
            if (wave == 1)
                tdm_load_2d(lb0 + (buf ^ 1) * bufBytes, Bblk + (s + 1) * GK,
                            GK, 64, (unsigned)K);
        }
#else
        {   // fallback: cooperative global->LDS copy (128 thr, 64B each row-half)
            int row  = threadIdx.x >> 1;
            int half = threadIdx.x & 1;
            const uint4* sa = (const uint4*)(Ablk + (size_t)row * K + s * GK + half * 32);
            const uint4* sb = (const uint4*)(Bblk + (size_t)row * K + s * GK + half * 32);
            uint4* da = (uint4*)(&ldsA[buf][row * GK + half * 32]);
            uint4* db = (uint4*)(&ldsB[buf][row * GK + half * 32]);
            da[0] = sa[0]; da[1] = sa[1]; da[2] = sa[2]; da[3] = sa[3];
            db[0] = sb[0]; db[1] = sb[1]; db[2] = sb[2]; db[3] = sb[3];
            __syncthreads();
        }
#endif
        // ---- compute on current buffer: 2 k-chunks x (1 A-frag, 4 B-frags) ----
        const bf16_t* ap = &ldsA[buf][wave * 16 * GK];
        const bf16_t* bp = &ldsB[buf][0];
#pragma unroll
        for (int kc = 0; kc < GK; kc += 32) {
            v16bf a = load_a_frag(ap + kc, GK, lane);
#pragma unroll
            for (int j = 0; j < 4; ++j) {
                v16bf b = load_b_frag(bp + j * 16 * GK + kc, GK, lane);
                acc[j] = WMMA_BF16(a, b, acc[j]);
            }
        }
#if USE_TDM
        tdm_wait();            // issuing waves: next panels landed
        __syncthreads();       // all waves: done reading buf, next buf visible
#else
        __syncthreads();
#endif
    }

    // C/D layout: lane holds col = lane%16, rows rbase+r, rbase = 8*(lane/16)
    int col   = lane & 15;
    int rbase = (lane >> 4) << 3;
#pragma unroll
    for (int j = 0; j < 4; ++j) {
        int n = bn * 64 + j * 16 + col;
#pragma unroll
        for (int r = 0; r < 8; ++r) {
            size_t m = (size_t)bm * 64 + wave * 16 + rbase + r;
            if (outf) outf[m * N + n] = acc[j][r];
            else      outb[m * N + n] = (bf16_t)acc[j][r];
        }
    }
}

// ---------------------------------------------------------------------------
// Kernel 3: RoPE + RMS-norm.  One wave per (b,h,t) row of D=128.
//   qkv : bf16 [B][T][3*C] ; qb,kb : [B][H][T][D] (q scaled by 1/sqrt(D))
//   vt  : bf16 [B][H][D][T] (transposed for PV B-fragments)
// ---------------------------------------------------------------------------
__global__ void k_rope_rms(const bf16_t* __restrict__ qkv,
                           bf16_t* __restrict__ qb,
                           bf16_t* __restrict__ kb,
                           bf16_t* __restrict__ vt) {
    int lane = threadIdx.x & 31;
    int wave = threadIdx.x >> 5;
    int gw   = blockIdx.x * 4 + wave;
    int t = gw % ATT_T;
    int h = (gw / ATT_T) % ATT_H;
    int b = gw / (ATT_T * ATT_H);

    size_t src = ((size_t)(b * ATT_T + t)) * (3 * ATT_C) + (size_t)h * ATT_D;
    int d0 = lane * 4;

    float qv[4], kv[4];
    bf16_t vraw[4];
#pragma unroll
    for (int i = 0; i < 4; ++i) {
        qv[i]   = (float)qkv[src             + d0 + i];
        kv[i]   = (float)qkv[src + ATT_C     + d0 + i];
        vraw[i] =        qkv[src + 2 * ATT_C + d0 + i];
    }

    float qo[4], ko[4];
#pragma unroll
    for (int p = 0; p < 2; ++p) {
        int j = 2 * lane + p;
        float freq = __powf(10000.0f, -((float)(2 * j)) / (float)ATT_D);
        float s, c;
        __sincosf((float)t * freq, &s, &c);
        qo[2*p]     = qv[2*p]     * c - qv[2*p + 1] * s;
        qo[2*p + 1] = qv[2*p + 1] * c + qv[2*p]     * s;
        ko[2*p]     = kv[2*p]     * c - kv[2*p + 1] * s;
        ko[2*p + 1] = kv[2*p + 1] * c + kv[2*p]     * s;
    }

    float sq = 0.f, sk = 0.f;
#pragma unroll
    for (int i = 0; i < 4; ++i) { sq += qo[i] * qo[i]; sk += ko[i] * ko[i]; }
#pragma unroll
    for (int m = 1; m <= 16; m <<= 1) {
        sq += __shfl_xor(sq, m);
        sk += __shfl_xor(sk, m);
    }
    float rq = 1.0f / __fsqrt_rn(sq * (1.0f / ATT_D) + 1e-8f);
    float rk = 1.0f / __fsqrt_rn(sk * (1.0f / ATT_D) + 1e-8f);
    rq *= 0.08838834764831845f;     // fold 1/sqrt(D) into q

    size_t dst   = ((size_t)((b * ATT_H + h) * ATT_T + t)) * ATT_D + d0;
    size_t vbase = ((size_t)(b * ATT_H + h)) * ATT_D * ATT_T;
#pragma unroll
    for (int i = 0; i < 4; ++i) {
        qb[dst + i] = (bf16_t)(qo[i] * rq);
        kb[dst + i] = (bf16_t)(ko[i] * rk);
        vt[vbase + (size_t)(d0 + i) * ATT_T + t] = vraw[i];
    }
}

// ---------------------------------------------------------------------------
// Kernel 4: causal flash attention. One wave per 16-query tile of one (b,h).
// ---------------------------------------------------------------------------
__global__ void __launch_bounds__(128)
k_flash_attn(const bf16_t* __restrict__ qb, const bf16_t* __restrict__ kbuf,
             const bf16_t* __restrict__ vt, bf16_t* __restrict__ yb) {
    __shared__ bf16_t plds_all[4 * 16 * 32];
    int lane = threadIdx.x & 31;
    int wave = threadIdx.x >> 5;
    bf16_t* lp = plds_all + wave * (16 * 32);

    int gw    = blockIdx.x * 4 + wave;
    int qtile = gw % (ATT_T / 16);
    int bh    = gw / (ATT_T / 16);
    int h     = bh % ATT_H;
    int b     = bh / ATT_H;
    int q0    = qtile * 16;

    const bf16_t* Q  = qb   + (size_t)bh * ATT_T * ATT_D;
    const bf16_t* Km = kbuf + (size_t)bh * ATT_T * ATT_D;
    const bf16_t* V  = vt   + (size_t)bh * ATT_D * ATT_T;

    v16bf qa[4];
#pragma unroll
    for (int c = 0; c < 4; ++c)
        qa[c] = load_a_frag(Q + (size_t)q0 * ATT_D + 32 * c, ATT_D, lane);

    v8f o[8];
#pragma unroll
    for (int j = 0; j < 8; ++j) o[j] = v8f{0};
    float mrun[8], lrun[8];
#pragma unroll
    for (int r = 0; r < 8; ++r) { mrun[r] = -1e30f; lrun[r] = 0.f; }

    int col   = lane & 15;
    int rbase = (lane >> 4) << 3;
    int kend  = q0 + 16;

    for (int kb0 = 0; kb0 < kend; kb0 += 32) {
        v8f s0 = v8f{0}, s1 = v8f{0};
#pragma unroll
        for (int c = 0; c < 4; ++c) {
            v16bf b0 = load_b_frag(Km + (size_t)kb0        * ATT_D + 32 * c, ATT_D, lane);
            v16bf b1 = load_b_frag(Km + (size_t)(kb0 + 16) * ATT_D + 32 * c, ATT_D, lane);
            s0 = WMMA_BF16(qa[c], b0, s0);
            s1 = WMMA_BF16(qa[c], b1, s1);
        }

#pragma unroll
        for (int r = 0; r < 8; ++r) {
            int qrow = q0 + rbase + r;
            float e0 = (kb0 + col      > qrow) ? -1e30f : s0[r];
            float e1 = (kb0 + 16 + col > qrow) ? -1e30f : s1[r];
            float vmax = fmaxf(e0, e1);
            vmax = fmaxf(vmax, __shfl_xor(vmax, 1));
            vmax = fmaxf(vmax, __shfl_xor(vmax, 2));
            vmax = fmaxf(vmax, __shfl_xor(vmax, 4));
            vmax = fmaxf(vmax, __shfl_xor(vmax, 8));
            float mnew = fmaxf(mrun[r], vmax);
            float corr = __expf(mrun[r] - mnew);
            float p0 = __expf(e0 - mnew);
            float p1 = __expf(e1 - mnew);
            float rs = p0 + p1;
            rs += __shfl_xor(rs, 1);
            rs += __shfl_xor(rs, 2);
            rs += __shfl_xor(rs, 4);
            rs += __shfl_xor(rs, 8);
            lrun[r] = lrun[r] * corr + rs;
            mrun[r] = mnew;
#pragma unroll
            for (int j = 0; j < 8; ++j) o[j][r] *= corr;
            lp[(rbase + r) * 32 + col]      = (bf16_t)p0;
            lp[(rbase + r) * 32 + col + 16] = (bf16_t)p1;
        }
        __asm__ volatile("" ::: "memory");

        v16bf pa = load_a_frag(lp, 32, lane);
#pragma unroll
        for (int j = 0; j < 8; ++j) {
            v16bf vb = load_b_frag(V + (size_t)(j * 16) * ATT_T + kb0, ATT_T, lane);
            o[j] = WMMA_BF16(pa, vb, o[j]);
        }
    }

#pragma unroll
    for (int r = 0; r < 8; ++r) {
        float inv = 1.0f / lrun[r];
        size_t row = (size_t)(b * ATT_T + q0 + rbase + r) * ATT_C + (size_t)h * ATT_D;
#pragma unroll
        for (int j = 0; j < 8; ++j)
            yb[row + j * 16 + col] = (bf16_t)(o[j][r] * inv);
    }
}

// ---------------------------------------------------------------------------
// Host-side launcher
// ---------------------------------------------------------------------------
extern "C" void kernel_launch(void* const* d_in, const int* in_sizes, int n_in,
                              void* d_out, int out_size, void* d_ws, size_t ws_size,
                              hipStream_t stream) {
    const float* x     = (const float*)d_in[0];
    const float* w_qkv = (const float*)d_in[1];
    const float* w_o   = (const float*)d_in[2];
    float* out = (float*)d_out;

    const size_t M  = (size_t)ATT_B * ATT_T;   // 8192
    const size_t C  = ATT_C;
    const size_t N3 = 3 * C;

    char* p = (char*)d_ws;
    bf16_t* xb    = (bf16_t*)p; p += M * C * 2;
    bf16_t* wqkvb = (bf16_t*)p; p += N3 * C * 2;
    bf16_t* wob   = (bf16_t*)p; p += C * C * 2;
    bf16_t* qkv   = (bf16_t*)p; p += M * N3 * 2;
    bf16_t* qb    = (bf16_t*)p; p += M * C * 2;
    bf16_t* kb    = (bf16_t*)p; p += M * C * 2;
    bf16_t* vt    = (bf16_t*)p; p += M * C * 2;
    bf16_t* yb    = (bf16_t*)p; p += M * C * 2;

    k_f32_to_bf16<<<2048, 256, 0, stream>>>(x,     xb,    M * C);
    k_f32_to_bf16<<<2048, 256, 0, stream>>>(w_qkv, wqkvb, N3 * C);
    k_f32_to_bf16<<<1024, 256, 0, stream>>>(w_o,   wob,   C * C);

    // QKV projection: [M,C] x [3C,C]^T -> [M,3C] bf16
    k_wmma_gemm<<<(int)((M / 64) * (N3 / 64)), 128, 0, stream>>>(
        xb, wqkvb, qkv, nullptr, (int)M, (int)N3, (int)C);

    // RoPE + RMS-norm + V transpose
    k_rope_rms<<<(ATT_B * ATT_H * ATT_T) / 4, 128, 0, stream>>>(qkv, qb, kb, vt);

    // causal flash attention
    k_flash_attn<<<(ATT_B * ATT_H * (ATT_T / 16)) / 4, 128, 0, stream>>>(qb, kb, vt, yb);

    // output projection: [M,C] x [C,C]^T -> f32 out
    k_wmma_gemm<<<(int)((M / 64) * (C / 64)), 128, 0, stream>>>(
        yb, wob, nullptr, out, (int)M, (int)C, (int)C);
}
